// TokenMixing_16260746183265
// MI455X (gfx1250) — matrix operand order repaired and checked by reading, not compiled, not run on hardware
//
#include <hip/hip_runtime.h>
#include <hip/hip_bf16.h>

typedef __attribute__((ext_vector_type(16))) _Float16 v16h;
typedef __attribute__((ext_vector_type(8)))  _Float16 v8h;
typedef __attribute__((ext_vector_type(8)))  float    v8f;
typedef __attribute__((ext_vector_type(4)))  int      v4i;

#define D_MODEL 1024
#define BATCH   8
#define SEQ     2048
#define MROWS   (BATCH * SEQ)   // 16384

#define BK 32                   // K-slice per stage
#define BSTRIDE 40              // LDS row stride in halfs (80B) -> conflict-free b128 reads

#if __has_builtin(__builtin_amdgcn_global_load_async_to_lds_b128) && \
    __has_builtin(__builtin_amdgcn_s_wait_asynccnt)
#define HAVE_ASYNC 1
#else
#define HAVE_ASYNC 0
#endif

#if HAVE_ASYNC
// global -> LDS async copy of 16B; imm byte offset applies to both sides (ISA §15.18)
#define GL_ASYNC_B128(gp, lp, off)                                        \
    __builtin_amdgcn_global_load_async_to_lds_b128(                       \
        (__attribute__((address_space(1))) v4i*)(gp),                     \
        (__attribute__((address_space(3))) v4i*)(lp), (off), 0)
#endif

__device__ __forceinline__ v8f wmma_f16(v16h a, v16h b, v8f c) {
    return __builtin_amdgcn_wmma_f32_16x16x32_f16(
        false, a, false, b, (short)0, c, false, false);
}

// ---------------- weights f32 -> f16 ----------------
__global__ __launch_bounds__(256)
void cvt_f16_kernel(const float* __restrict__ src, _Float16* __restrict__ dst, int n) {
    int i = blockIdx.x * blockDim.x + threadIdx.x;
    if (i < n) dst[i] = (_Float16)src[i];
}

// ---------------- token-shift mixing -> f16 ----------------
__global__ __launch_bounds__(256)
void prep_kernel(const float* __restrict__ x,
                 const float* __restrict__ tmk, const float* __restrict__ tmv,
                 const float* __restrict__ tmr,
                 _Float16* __restrict__ xk, _Float16* __restrict__ xv,
                 _Float16* __restrict__ xr)
{
    int c = blockIdx.x * blockDim.x + threadIdx.x;   // 0..1023
    int m = blockIdx.y;                              // 0..16383 (b*T+t)
    int t = m & (SEQ - 1);
    size_t idx = (size_t)m * D_MODEL + c;
    float xc = x[idx];
    float xs = (t == 0) ? 0.f : x[idx - D_MODEL];
    float mk = tmk[c], mv = tmv[c], mr = tmr[c];
    xk[idx] = (_Float16)(xc * mk + xs * (1.f - mk));
    xv[idx] = (_Float16)(xc * mv + xs * (1.f - mv));
    xr[idx] = (_Float16)(xc * mr + xs * (1.f - mr));
}

// Load both A fragments (rows m0..m0+31) for one 32-wide K step.
__device__ __forceinline__ void load_a(const _Float16* __restrict__ Ar0,
                                       const _Float16* __restrict__ Ar1,
                                       int kk, int half, v16h& a0, v16h& a1)
{
    v8h a0lo = *(const v8h*)(Ar0 + kk + half * 16);
    v8h a0hi = *(const v8h*)(Ar0 + kk + half * 16 + 8);
    v8h a1lo = *(const v8h*)(Ar1 + kk + half * 16);
    v8h a1hi = *(const v8h*)(Ar1 + kk + half * 16 + 8);
#pragma unroll
    for (int i = 0; i < 8; ++i) {
        a0[i] = a0lo[i]; a0[i + 8] = a0hi[i];
        a1[i] = a1lo[i]; a1[i + 8] = a1hi[i];
    }
}

// One 32-wide K step for a 32(M)x128(N) wave tile; B fragments read from LDS,
// software-pipelined so ds_load of nt+1 overlaps the WMMAs of nt.
__device__ __forceinline__ void compute_step(const v16h& a0, const v16h& a1,
                                             const _Float16* __restrict__ Bbuf,
                                             int half, int l15,
                                             v8f acc[2][8])
{
    const _Float16* Br0 = Bbuf + (size_t)l15 * BSTRIDE + half * 16;
    v8h blo = *(const v8h*)(Br0);
    v8h bhi = *(const v8h*)(Br0 + 8);
#pragma unroll
    for (int nt = 0; nt < 8; ++nt) {
        v8h nlo = blo, nhi = bhi;
        if (nt < 7) {
            const _Float16* Br = Bbuf + (size_t)((nt + 1) * 16 + l15) * BSTRIDE + half * 16;
            nlo = *(const v8h*)(Br);
            nhi = *(const v8h*)(Br + 8);
        }
        v16h bf;
#pragma unroll
        for (int i = 0; i < 8; ++i) { bf[i] = blo[i]; bf[i + 8] = bhi[i]; }
        acc[0][nt] = wmma_f16(a0, bf, acc[0][nt]);
        acc[1][nt] = wmma_f16(a1, bf, acc[1][nt]);
        blo = nlo; bhi = nhi;
    }
}

// ---------------- WMMA GEMM: C[M,N] f32 = A[M,K] f16 (rowmajor) * W[N,K]^T f16 (rowmajor)
// Block: 256 threads = 8 waves. Wave tile: 32(M) x 128(N). Block tile: 256 x 128.
// B slice (128 x 32) staged in LDS (async global->LDS, double-buffered);
// A fragments prefetched one K step ahead in registers.
__global__ __launch_bounds__(256)
void gemm_wmma_kernel(const _Float16* __restrict__ A,
                      const _Float16* __restrict__ W,
                      float* __restrict__ C,
                      int M, int N, int K)
{
    __shared__ _Float16 Bs[2][128 * BSTRIDE];   // 2 x 10KB

    const int tid  = threadIdx.x;
    const int lane = tid & 31;
    const int wave = tid >> 5;
    const int half = lane >> 4;
    const int l15  = lane & 15;

    const int m0 = blockIdx.x * 256 + wave * 32;
    const int n0 = blockIdx.y * 128;

    // staging role: 2 threads per B row, 32B (16 halfs) each
    const int srow  = tid >> 1;      // 0..127
    const int spart = tid & 1;       // 0..1
    const _Float16* __restrict__ Wsrc = W + (size_t)(n0 + srow) * K + spart * 16;
    _Float16* const sdst0 = &Bs[0][srow * BSTRIDE + spart * 16];
    _Float16* const sdst1 = &Bs[1][srow * BSTRIDE + spart * 16];

    v8f acc[2][8] = {};
    const _Float16* __restrict__ Ar0 = A + (size_t)(m0 + l15) * K;
    const _Float16* __restrict__ Ar1 = Ar0 + (size_t)16 * K;

#if HAVE_ASYNC
    // prologue: stage B(k=0) into buf 0, prefetch A(k=0) into registers
    GL_ASYNC_B128(Wsrc, sdst0, 0);
    GL_ASYNC_B128(Wsrc, sdst0, 16);
    v16h a0, a1;
    load_a(Ar0, Ar1, 0, half, a0, a1);

    int p = 0;
    for (int kk = 0; kk < K; kk += BK, p ^= 1) {
        if (kk + BK < K) {
            const _Float16* g = Wsrc + kk + BK;
            _Float16* l = p ? sdst0 : sdst1;
            GL_ASYNC_B128(g, l, 0);
            GL_ASYNC_B128(g, l, 16);
            __builtin_amdgcn_s_wait_asynccnt(2);   // current slice landed in LDS
        } else {
            __builtin_amdgcn_s_wait_asynccnt(0);
        }
        __syncthreads();                            // all waves' slices visible

        v16h a0n = a0, a1n = a1;
        if (kk + BK < K) load_a(Ar0, Ar1, kk + BK, half, a0n, a1n);  // next-step A prefetch

        compute_step(a0, a1, &Bs[p][0], half, l15, acc);
        __syncthreads();                            // done reading before next overwrite
        a0 = a0n; a1 = a1n;
    }
#else
    for (int kk = 0; kk < K; kk += BK) {
        __syncthreads();
        v8h t0 = *(const v8h*)(Wsrc + kk);
        v8h t1 = *(const v8h*)(Wsrc + kk + 8);
        *(v8h*)(sdst0)     = t0;
        *(v8h*)(sdst0 + 8) = t1;
        __syncthreads();
        v16h a0, a1;
        load_a(Ar0, Ar1, kk, half, a0, a1);
        compute_step(a0, a1, &Bs[0][0], half, l15, acc);
    }
#endif

    // C/D layout: VGPR r: lanes 0-15 -> M = r, lanes 16-31 -> M = 8 + r; N = l15
#pragma unroll
    for (int mi = 0; mi < 2; ++mi)
#pragma unroll
        for (int nt = 0; nt < 8; ++nt)
#pragma unroll
            for (int r = 0; r < 8; ++r)
                C[(size_t)(m0 + mi * 16 + half * 8 + r) * N + (n0 + nt * 16 + l15)]
                    = acc[mi][nt][r];
}

// ---------------- numerically-stable WKV scan + sigmoid(r), emit f16 ----------------
__global__ __launch_bounds__(256)
void wkv_kernel(const float* __restrict__ kbuf, const float* __restrict__ vbuf,
                const float* __restrict__ rbuf,
                const float* __restrict__ w, const float* __restrict__ u,
                _Float16* __restrict__ rwkv)
{
    int gid = blockIdx.x * blockDim.x + threadIdx.x;  // 0..8191
    int c = gid & (D_MODEL - 1);
    int b = gid >> 10;
    float wn = -expf(w[c]);
    float uc = u[c];
    float aa = 0.f, bb = 0.f, pp = -1e38f;
    size_t idx = (size_t)b * SEQ * D_MODEL + c;
    for (int t = 0; t < SEQ; ++t, idx += D_MODEL) {
        float kt = kbuf[idx];
        float vt = vbuf[idx];
        float ww = uc + kt;
        float p  = fmaxf(pp, ww);
        float e1 = expf(pp - p);
        float e2 = expf(ww - p);
        float y  = (e1 * aa + e2 * vt) / (e1 * bb + e2);
        float rv  = rbuf[idx];
        float sig = 1.f / (1.f + expf(-rv));
        rwkv[idx] = (_Float16)(sig * y);
        float ww2 = pp + wn;
        float p2  = fmaxf(ww2, kt);
        float e1b = expf(ww2 - p2);
        float e2b = expf(kt - p2);
        aa = e1b * aa + e2b * vt;
        bb = e1b * bb + e2b;
        pp = p2;
    }
}

extern "C" void kernel_launch(void* const* d_in, const int* in_sizes, int n_in,
                              void* d_out, int out_size, void* d_ws, size_t ws_size,
                              hipStream_t stream)
{
    const float* x   = (const float*)d_in[0];
    const float* w   = (const float*)d_in[1];
    const float* u   = (const float*)d_in[2];
    const float* tmk = (const float*)d_in[3];
    const float* tmv = (const float*)d_in[4];
    const float* tmr = (const float*)d_in[5];
    const float* Wk  = (const float*)d_in[6];
    const float* Wv  = (const float*)d_in[7];
    const float* Wr  = (const float*)d_in[8];
    const float* Wo  = (const float*)d_in[9];
    float* out = (float*)d_out;

    const size_t WMAT = (size_t)D_MODEL * D_MODEL;   // 1M elements
    const size_t MC   = (size_t)MROWS * D_MODEL;     // 16.7M elements

    _Float16* Wk16 = (_Float16*)d_ws;
    _Float16* Wv16 = Wk16 + WMAT;
    _Float16* Wr16 = Wv16 + WMAT;
    _Float16* Wo16 = Wr16 + WMAT;
    _Float16* xk16 = Wo16 + WMAT;        // 32MB; reused later for rwkv16
    _Float16* xv16 = xk16 + MC;
    _Float16* xr16 = xv16 + MC;
    float*    kf   = (float*)(xr16 + MC);
    float*    vf   = kf + MC;
    float*    rf   = vf + MC;

    // 1) weights -> f16
    const int wn = (int)WMAT;
    cvt_f16_kernel<<<(wn + 255) / 256, 256, 0, stream>>>(Wk, Wk16, wn);
    cvt_f16_kernel<<<(wn + 255) / 256, 256, 0, stream>>>(Wv, Wv16, wn);
    cvt_f16_kernel<<<(wn + 255) / 256, 256, 0, stream>>>(Wr, Wr16, wn);
    cvt_f16_kernel<<<(wn + 255) / 256, 256, 0, stream>>>(Wo, Wo16, wn);

    // 2) token-shift mixing
    prep_kernel<<<dim3(D_MODEL / 256, MROWS), 256, 0, stream>>>(
        x, tmk, tmv, tmr, xk16, xv16, xr16);

    // 3) k/v/r projections (WMMA)
    dim3 gg(MROWS / 256, D_MODEL / 128);
    gemm_wmma_kernel<<<gg, 256, 0, stream>>>(xk16, Wk16, kf, MROWS, D_MODEL, D_MODEL);
    gemm_wmma_kernel<<<gg, 256, 0, stream>>>(xv16, Wv16, vf, MROWS, D_MODEL, D_MODEL);
    gemm_wmma_kernel<<<gg, 256, 0, stream>>>(xr16, Wr16, rf, MROWS, D_MODEL, D_MODEL);

    // 4) WKV scan + sigmoid gating; rwkv (f16) overwrites dead xk16 buffer
    wkv_kernel<<<(BATCH * D_MODEL) / 256, 256, 0, stream>>>(kf, vf, rf, w, u, xk16);

    // 5) output projection (WMMA) -> f32 out
    gemm_wmma_kernel<<<gg, 256, 0, stream>>>(xk16, Wo16, out, MROWS, D_MODEL, D_MODEL);
}
